// SiRNAEfficiencyPredictorGNN_40510131536343
// MI455X (gfx1250) — matrix operand order repaired and checked by reading, not compiled, not run on hardware
//
#include <hip/hip_runtime.h>

#define NODES 20000
#define EDGES 320000
#define OUTC  256
#define HD    128

typedef __attribute__((ext_vector_type(16))) __bf16 bf16x16;
typedef __attribute__((ext_vector_type(8)))  float  f32x8;

__device__ __forceinline__ unsigned short f2bf(float f) {
    unsigned int u = __float_as_uint(f);
    u += 0x7FFFu + ((u >> 16) & 1u);       // round-to-nearest-even bf16
    return (unsigned short)(u >> 16);
}

union FragB16 { uint4 q[2]; bf16x16 v; };

// ---------------- conversion kernels ----------------

__global__ void k_cvt_edges(const long long* __restrict__ ei64, int* __restrict__ ei32) {
    int t = blockIdx.x * blockDim.x + threadIdx.x;
    if (t < 2 * EDGES) ei32[t] = (int)ei64[t];
}

__global__ void k_feat2bf(const float* __restrict__ x, unsigned short* __restrict__ hb,
                          int n_elems, int din, int kpad) {
    int t = blockIdx.x * blockDim.x + threadIdx.x;
    if (t >= n_elems) return;
    int n = t / kpad, kk = t % kpad;
    hb[t] = (kk < din) ? f2bf(x[n * din + kk]) : (unsigned short)0;
}

// W [din, 256] f32 (row-major) -> Wt [256, kpad] bf16 (n-major, k-contiguous), zero-padded K
__global__ void k_wconvT(const float* __restrict__ W, unsigned short* __restrict__ Wt,
                         int din, int kpad) {
    int t = blockIdx.x * blockDim.x + threadIdx.x;
    if (t >= OUTC * kpad) return;
    int n = t / kpad, kk = t % kpad;
    Wt[t] = (kk < din) ? f2bf(W[kk * OUTC + n]) : (unsigned short)0;
}

// ---------------- WMMA GEMM: C[N,256] = A_bf16[N,kpad] * B_bf16[kpad,256] + bias ----------------
// Grid: 5000 blocks x 128 threads (4 waves); one 16x16 tile per wave; 1250 m-tiles x 16 n-tiles.

__global__ void k_gemm_wmma(const unsigned short* __restrict__ A,
                            const unsigned short* __restrict__ Bt,   // [256, kpad] (B transposed)
                            const float* __restrict__ bias,
                            float* __restrict__ C, int kpad) {
    int wave = threadIdx.x >> 5;
    int lane = threadIdx.x & 31;
    int tile = blockIdx.x * 4 + wave;
    int mtile = tile >> 4;          // 0..1249
    int ntile = tile & 15;          // 0..15
    int m0 = mtile * 16, n0 = ntile * 16;
    int half = lane >> 4;
    int l15  = lane & 15;

    const unsigned short* Arow = A  + (size_t)(m0 + l15) * kpad;   // A row m, contiguous K
    const unsigned short* Brow = Bt + (size_t)(n0 + l15) * kpad;   // B col n, contiguous K

    f32x8 acc = {};
    for (int k0 = 0; k0 < kpad; k0 += 32) {
        FragB16 af, bfrag;
        // A 16-bit layout: elems 0..7 <-> K = k0 + half*8 + 0..7 ; elems 8..15 <-> K = k0+16+half*8+0..7
        af.q[0] = *(const uint4*)(Arow + k0 + half * 8);
        af.q[1] = *(const uint4*)(Arow + k0 + 16 + half * 8);
        // B 16-bit layout: elems 0..7 <-> K = k0 + half*8 + 0..7 ; elems 8..15 <-> K = k0+16+half*8+0..7
        bfrag.q[0] = *(const uint4*)(Brow + k0 + half * 8);
        bfrag.q[1] = *(const uint4*)(Brow + k0 + 16 + half * 8);
        acc = __builtin_amdgcn_wmma_f32_16x16x32_bf16(false, af.v, false, bfrag.v,
                                                      (short)0, acc, false, false);
    }
    int col = n0 + l15;
    float bv = bias[col];
    float* Cp = C + (size_t)col;
#pragma unroll
    for (int r = 0; r < 8; ++r) {
        int m = m0 + r + half * 8;          // C/D layout: vgpr r, lane-half -> row
        Cp[(size_t)m * OUTC] = acc[r] + bv;
    }
}

// ---------------- attention kernels ----------------

__global__ void k_init_ms(float* __restrict__ mrow, float* __restrict__ srow) {
    int t = blockIdx.x * blockDim.x + threadIdx.x;
    if (t < NODES * 2) { mrow[t] = -INFINITY; srow[t] = 0.0f; }
}

__device__ __forceinline__ void atomicMaxF(float* addr, float value) {
    if (value >= 0.0f) atomicMax((int*)addr, __float_as_int(value));
    else               atomicMin((unsigned int*)addr, __float_as_uint(value));
}

// one block (64 thr = 2 waves) per edge; wave = head; lane-parallel dot of length 128
__global__ void k_edge_logits(const float* __restrict__ q, const float* __restrict__ k,
                              const int* __restrict__ srci, const int* __restrict__ dsti,
                              float* __restrict__ logits, float* __restrict__ mrow) {
    int e = blockIdx.x;
    int h = threadIdx.x >> 5;
    int lane = threadIdx.x & 31;
    int s = srci[e], d = dsti[e];
    const float* qp = q + (size_t)d * OUTC + h * HD;
    const float* kp = k + (size_t)s * OUTC + h * HD;
    float acc = 0.0f;
#pragma unroll
    for (int i = 0; i < 4; ++i) {
        int idx = lane + i * 32;
        acc += qp[idx] * kp[idx];
    }
#pragma unroll
    for (int m = 16; m >= 1; m >>= 1) acc += __shfl_xor(acc, m, 32);
    if (lane == 0) {
        acc *= 0.08838834764831845f;      // 1/sqrt(128)
        logits[(size_t)e * 2 + h] = acc;
        atomicMaxF(&mrow[(size_t)d * 2 + h], acc);
    }
}

__global__ void k_edge_exp(float* __restrict__ logits, const int* __restrict__ dsti,
                           const float* __restrict__ mrow, float* __restrict__ srow) {
    int t = blockIdx.x * blockDim.x + threadIdx.x;
    if (t >= EDGES * 2) return;
    int e = t >> 1, h = t & 1;
    int d = dsti[e];
    float ex = __expf(logits[t] - mrow[(size_t)d * 2 + h]);
    logits[t] = ex;
    atomicAdd(&srow[(size_t)d * 2 + h], ex);
}

// one block (256 thr) per edge: thread = (head, d-channel); scatter alpha * v[src] into agg[dst]
__global__ void k_edge_agg(const float* __restrict__ logits, const float* __restrict__ srow,
                           const float* __restrict__ v, const int* __restrict__ srci,
                           const int* __restrict__ dsti, float* __restrict__ agg) {
    int e = blockIdx.x;
    int t = threadIdx.x;
    int h = t >> 7, di = t & 127;
    int sN = srci[e], dN = dsti[e];
    float alpha = logits[(size_t)e * 2 + h] / srow[(size_t)dN * 2 + h];
    atomicAdd(&agg[(size_t)dN * OUTC + h * HD + di],
              alpha * v[(size_t)sN * OUTC + h * HD + di]);
}

// ---------------- relu + layernorm (+ bf16 re-emit for next layer) ----------------

__global__ void k_relu_ln(const float* __restrict__ in, const float* __restrict__ g,
                          const float* __restrict__ be, float* __restrict__ out,
                          unsigned short* __restrict__ hb) {
    __shared__ float red[256];
    int n = blockIdx.x, t = threadIdx.x;
    float v = fmaxf(in[(size_t)n * OUTC + t], 0.0f);
    red[t] = v;
    __syncthreads();
    for (int s = 128; s > 0; s >>= 1) { if (t < s) red[t] += red[t + s]; __syncthreads(); }
    float mu = red[0] * (1.0f / 256.0f);
    __syncthreads();
    float dv = v - mu;
    red[t] = dv * dv;
    __syncthreads();
    for (int s = 128; s > 0; s >>= 1) { if (t < s) red[t] += red[t + s]; __syncthreads(); }
    float var = red[0] * (1.0f / 256.0f);
    float y = dv * rsqrtf(var + 1e-5f) * g[t] + be[t];
    out[(size_t)n * OUTC + t] = y;
    hb[(size_t)n * OUTC + t] = f2bf(y);
}

// ---------------- pooling + final MLP ----------------

__global__ void k_colmean(const float* __restrict__ h, float* __restrict__ pooled) {
    __shared__ float red[256];
    int c = blockIdx.x, t = threadIdx.x;
    float s = 0.0f;
    for (int n = t; n < NODES; n += 256) s += h[(size_t)n * OUTC + c];
    red[t] = s;
    __syncthreads();
    for (int k = 128; k > 0; k >>= 1) { if (t < k) red[t] += red[t + k]; __syncthreads(); }
    if (t == 0) pooled[c] = red[0] * (1.0f / (float)NODES);
}

__global__ void k_final(const float* __restrict__ pooled, const float* __restrict__ Wfc1,
                        const float* __restrict__ bfc1, const float* __restrict__ Wfc2,
                        const float* __restrict__ bfc2, float* __restrict__ out) {
    __shared__ float red[128];
    int t = threadIdx.x;
    float acc = bfc1[t];
    for (int i = 0; i < OUTC; ++i) acc += pooled[i] * Wfc1[i * HD + t];
    acc = fmaxf(acc, 0.0f);
    red[t] = acc * Wfc2[t];
    __syncthreads();
    for (int s = 64; s > 0; s >>= 1) { if (t < s) red[t] += red[t + s]; __syncthreads(); }
    if (t == 0) out[0] = 1.0f / (1.0f + __expf(-(red[0] + bfc2[0])));
}

// ---------------- host orchestration ----------------

extern "C" void kernel_launch(void* const* d_in, const int* in_sizes, int n_in,
                              void* d_out, int out_size, void* d_ws, size_t ws_size,
                              hipStream_t stream) {
    (void)in_sizes; (void)n_in; (void)out_size; (void)ws_size;
    const float*     x  = (const float*)d_in[0];
    const long long* ei = (const long long*)d_in[1];
    // params: d_in[2 + 10*l + j], j: 0 Wq,1 bq,2 Wk,3 bk,4 Wv,5 bv,6 Ws,7 bs,8 g,9 be
    auto P = [&](int i) { return (const float*)d_in[2 + i]; };

    char* base = (char*)d_ws;
    size_t off = 0;
    auto alloc = [&](size_t bytes) -> void* {
        void* p = base + off;
        off = (off + bytes + 255) & ~(size_t)255;
        return p;
    };
    float* hcur = (float*)alloc(sizeof(float) * NODES * OUTC);
    float* agg  = (float*)alloc(sizeof(float) * NODES * OUTC);
    float* qb   = (float*)alloc(sizeof(float) * NODES * OUTC);
    float* kb   = (float*)alloc(sizeof(float) * NODES * OUTC);
    float* vb   = (float*)alloc(sizeof(float) * NODES * OUTC);
    unsigned short* hb = (unsigned short*)alloc(2ULL * NODES * OUTC);
    unsigned short* Wt = (unsigned short*)alloc(2ULL * 4 * OUTC * OUTC);
    float* logits = (float*)alloc(sizeof(float) * EDGES * 2);
    float* mrow   = (float*)alloc(sizeof(float) * NODES * 2);
    float* srow   = (float*)alloc(sizeof(float) * NODES * 2);
    int*   ei32   = (int*)alloc(sizeof(int) * 2 * EDGES);
    float* pooled = (float*)alloc(sizeof(float) * OUTC);

    k_cvt_edges<<<(2 * EDGES + 255) / 256, 256, 0, stream>>>(ei, ei32);
    const int* srci = ei32;
    const int* dsti = ei32 + EDGES;

    for (int l = 0; l < 3; ++l) {
        int din  = (l == 0) ? 4  : 256;
        int kpad = (l == 0) ? 32 : 256;
        const float* Wm[4]   = { P(10 * l + 0), P(10 * l + 2), P(10 * l + 4), P(10 * l + 6) };
        const float* bm[4]   = { P(10 * l + 1), P(10 * l + 3), P(10 * l + 5), P(10 * l + 7) };
        const float* g  = P(10 * l + 8);
        const float* be = P(10 * l + 9);

        if (l == 0)
            k_feat2bf<<<(NODES * 32 + 255) / 256, 256, 0, stream>>>(x, hb, NODES * 32, 4, 32);

        unsigned short* Wts[4] = { Wt, Wt + OUTC * 256, Wt + 2 * OUTC * 256, Wt + 3 * OUTC * 256 };
        for (int j = 0; j < 4; ++j)
            k_wconvT<<<(OUTC * kpad + 255) / 256, 256, 0, stream>>>(Wm[j], Wts[j], din, kpad);

        k_gemm_wmma<<<5000, 128, 0, stream>>>(hb, Wts[0], bm[0], qb,  kpad);
        k_gemm_wmma<<<5000, 128, 0, stream>>>(hb, Wts[1], bm[1], kb,  kpad);
        k_gemm_wmma<<<5000, 128, 0, stream>>>(hb, Wts[2], bm[2], vb,  kpad);
        k_gemm_wmma<<<5000, 128, 0, stream>>>(hb, Wts[3], bm[3], agg, kpad);  // skip connection

        k_init_ms<<<(NODES * 2 + 255) / 256, 256, 0, stream>>>(mrow, srow);
        k_edge_logits<<<EDGES, 64, 0, stream>>>(qb, kb, srci, dsti, logits, mrow);
        k_edge_exp<<<(EDGES * 2 + 255) / 256, 256, 0, stream>>>(logits, dsti, mrow, srow);
        k_edge_agg<<<EDGES, 256, 0, stream>>>(logits, srow, vb, srci, dsti, agg);

        k_relu_ln<<<NODES, 256, 0, stream>>>(agg, g, be, hcur, hb);
    }

    k_colmean<<<OUTC, 256, 0, stream>>>(hcur, pooled);
    k_final<<<1, 128, 0, stream>>>(pooled, (const float*)d_in[32], (const float*)d_in[33],
                                   (const float*)d_in[34], (const float*)d_in[35],
                                   (float*)d_out);
}
